// ElmoEmbeddings_33114197852765
// MI455X (gfx1250) — compile-verified, matrix operand order checked by reading, not run on hardware
//
#include <hip/hip_runtime.h>
#include <hip/hip_bf16.h>
#include <stdint.h>

// ---------------- problem constants ----------------
static constexpr int B_ = 8;
static constexpr int T_ = 96;
static constexpr int E_ = 256;
static constexpr int H_ = 2048;
static constexpr int P_ = 256;
static constexpr int NWG = 8;            // workgroups per direction (K/N split of gates)
static constexpr float CELL_CLIP = 3.0f;
static constexpr float PROJ_CLIP = 3.0f;

typedef __attribute__((ext_vector_type(16))) __bf16 v16bf;
typedef __attribute__((ext_vector_type(8)))  float  v8f;

// ---------------- embedding gather ----------------
__global__ void gather_kernel(const int* __restrict__ tokens,
                              const float* __restrict__ table,
                              float* __restrict__ emb) {
  int idx = blockIdx.x * 256 + threadIdx.x;        // < B*T*E (exact multiple)
  int e  = idx & 255;
  int bt = idx >> 8;
  int tok = tokens[bt];
  emb[idx] = table[(size_t)tok * E_ + e];
}

// K index map shared by A/B 16-bit WMMA operand layouts (ISA 7.12.2)
__device__ inline int kmap(int kh, int e) {
  int v = e >> 1;
  return (v < 4) ? (kh + 2 * v + (e & 1)) : (16 + kh + 2 * (v - 4) + (e & 1));
}

// ---------------- weight repack: f32 row-major (K x N) -> bf16 WMMA B-operand tiles --------
// dst layout: [mat][ntile][kblock][lane(32)][16 bf16]  (one 32x16 B tile = 1KB contiguous)
__global__ void pack_w_kernel(const float* __restrict__ src, __bf16* __restrict__ dst,
                              int K, int N, long total) {
  long idx = (long)blockIdx.x * blockDim.x + threadIdx.x;
  if (idx >= total) return;
  int  KB  = K >> 5;
  long per = (long)K * N;
  int  mat = (int)(idx / per);
  long r   = idx - (long)mat * per;
  int  e    = (int)(r & 15);
  int  lane = (int)((r >> 4) & 31);
  long tile = r >> 9;
  int  kb   = (int)(tile % KB);
  int  nt   = (int)(tile / KB);
  int  n    = (nt << 4) + (lane & 15);
  int  kh   = (lane >> 4) << 3;
  int  k    = (kb << 5) + kmap(kh, e);
  dst[idx] = (__bf16)src[((long)mat * K + k) * N + n];
}

// ---------------- cross-workgroup sync (8 co-resident WGs per direction) ----------------
__device__ inline void grid_sync(unsigned* c, unsigned target) {
  __threadfence();
  __syncthreads();
  if (threadIdx.x == 0) {
    __hip_atomic_fetch_add(c, 1u, __ATOMIC_ACQ_REL, __HIP_MEMORY_SCOPE_AGENT);
    while (__hip_atomic_load(c, __ATOMIC_ACQUIRE, __HIP_MEMORY_SCOPE_AGENT) < target)
      __builtin_amdgcn_s_sleep(1);
  }
  __syncthreads();
  __threadfence();
}

// Build one A-operand slot (kb = tid>>5, lane = tid&31) from LDS f32 [8][256] source.
// Rows 8..15 (batch padding) are zero.
__device__ inline void fill_a_from_lds(__bf16* __restrict__ dstA,
                                       const float* __restrict__ src) {
  const int s = threadIdx.x;            // exactly 256 slots
  const int kb = s >> 5, lane = s & 31;
  const int row = lane & 15;
  const int kh = (lane >> 4) << 3;
  v16bf a;
#pragma unroll
  for (int e = 0; e < 16; ++e) {
    int k = (kb << 5) + kmap(kh, e);
    a[e] = (row < 8) ? (__bf16)src[row * 256 + k] : (__bf16)0.f;
  }
  *(v16bf*)(dstA + s * 16) = a;
}

// Same, but sourcing x_t rows directly from global X[b][t][256].
__device__ inline void fill_a_from_global(__bf16* __restrict__ dstA,
                                          const float* __restrict__ X, int tt) {
  const int s = threadIdx.x;
  const int kb = s >> 5, lane = s & 31;
  const int row = lane & 15;
  const int kh = (lane >> 4) << 3;
  const float* rp = X + ((size_t)row * T_ + tt) * P_;
  v16bf a;
#pragma unroll
  for (int e = 0; e < 16; ++e) {
    int k = (kb << 5) + kmap(kh, e);
    a[e] = (row < 8) ? (__bf16)rp[k] : (__bf16)0.f;
  }
  *(v16bf*)(dstA + s * 16) = a;
}

// ---------------- persistent recurrent LSTM-projection kernel ----------------
__global__ __launch_bounds__(256) void lstm_kernel(
    const float*  __restrict__ emb,
    const __bf16* __restrict__ wkt,
    const __bf16* __restrict__ wrt,
    const __bf16* __restrict__ wpt,
    const float*  __restrict__ bias_all,
    const float*  __restrict__ mask,
    float*        __restrict__ Obuf,      // [4 slots = dir*2+layer][B][T][P] f32
    float*        __restrict__ hpart,     // [parity2][dir2][wg8][16][256] f32
    unsigned*     __restrict__ cnt)       // [2] sync counters
{
  // A-operand tiles in WMMA layout: [kb(8)][lane(32)][16 bf16] = 8KB each
  __shared__ __attribute__((aligned(32))) __bf16 xa[4096];
  __shared__ __attribute__((aligned(32))) __bf16 ha[4096];
  __shared__ __attribute__((aligned(32))) __bf16 hta[4096];
  __shared__ float h_f32[2048];   // [8][256] previous h (for mask fallback / A rebuild)
  __shared__ float ht_f32[2048];  // [8][256] o*tanh(c)
  __shared__ float m_lds[8];

  const int wg   = blockIdx.x;
  const int d    = blockIdx.y;
  const int tid  = threadIdx.x;
  const int lane = tid & 31;
  const int w    = tid >> 5;
  const int hw_local = w << 5;               // wave's local H base in [0,256)
  const int Hw   = (wg << 8) + hw_local;     // global H base for gate columns
  const int nlo  = lane & 15;
  const int rowsel = (lane >> 4) << 3;       // C-layout row offset (0 or 8)

  unsigned epoch = 0;
  int sidx = 0;
  const bool rev = (d == 1);

  for (int l = 0; l < 2; ++l) {
    const int matm = d * 2 + l;
    const __bf16* wk = wkt + (size_t)matm * (256 * 8192);
    const __bf16* wr = wrt + (size_t)matm * (256 * 8192);
    const __bf16* wp = wpt + (size_t)matm * (2048 * 256);
    const float* bias = bias_all + (size_t)matm * 8192;
    const float* X = (l == 0) ? emb : (Obuf + (size_t)(d * 2) * (B_ * T_ * P_));
    float* Oout = Obuf + (size_t)(d * 2 + l) * (B_ * T_ * P_);

    // per-lane bias values, hoisted once per layer (added post-GEMM in gate phase)
    float bv[8];
#pragma unroll
    for (int g = 0; g < 4; ++g)
#pragma unroll
      for (int th = 0; th < 2; ++th)
        bv[g * 2 + th] = bias[g * 2048 + Hw + th * 16 + nlo];

    // reset recurrent state
    for (int i = tid; i < 2048; i += 256) h_f32[i] = 0.f;
    {
      v16bf z;
#pragma unroll
      for (int e = 0; e < 16; ++e) z[e] = (__bf16)0.f;
      *(v16bf*)(ha + tid * 16) = z;
    }
    v8f cst[2];
#pragma unroll
    for (int th = 0; th < 2; ++th)
#pragma unroll
      for (int v = 0; v < 8; ++v) cst[th][v] = 0.f;
    __syncthreads();

    for (int t = 0; t < T_; ++t) {
      const int tt = rev ? (T_ - 1 - t) : t;

      // Defeat LICM while keeping global-address-space provenance:
      // add a laundered zero offset so weight loads stay in-loop as
      // global_load_b128 (SGPR base + lane voffset + imm tile offset),
      // re-streamed from L2 every step.
      size_t z0 = 0, z1 = 0, z2 = 0;
      asm volatile("" : "+s"(z0), "+s"(z1), "+s"(z2));
      const __bf16* wk_t = wk + z0;
      const __bf16* wr_t = wr + z1;
      const __bf16* wp_t = wp + z2;

      // ---- stage A tiles for x_t (bf16, WMMA layout) + per-step mask ----
      fill_a_from_global(xa, X, tt);
      if (tid < 8) m_lds[tid] = mask[tid * T_ + tt];
      __syncthreads();

      // ---- z = x@Wk + h@Wr : 8 C tiles (4 gates x 2 n-tiles) per wave ----
      v8f acc[8];
#pragma unroll
      for (int j = 0; j < 8; ++j)
#pragma unroll
        for (int v = 0; v < 8; ++v) acc[j][v] = 0.f;   // folds to inline SRC2=0

      for (int kb = 0; kb < 8; ++kb) {
        v16bf a = *(const v16bf*)(xa + (kb * 32 + lane) * 16);
#pragma unroll
        for (int g = 0; g < 4; ++g)
#pragma unroll
          for (int th = 0; th < 2; ++th) {
            int nt = g * 128 + (Hw >> 4) + th;
            v16bf bm = *(const v16bf*)(wk_t + ((size_t)(nt * 8 + kb) * 32 + lane) * 16);
            acc[g * 2 + th] = __builtin_amdgcn_wmma_f32_16x16x32_bf16(
                false, a, false, bm, (short)0, acc[g * 2 + th], false, false);
          }
      }
      for (int kb = 0; kb < 8; ++kb) {
        v16bf a = *(const v16bf*)(ha + (kb * 32 + lane) * 16);
#pragma unroll
        for (int g = 0; g < 4; ++g)
#pragma unroll
          for (int th = 0; th < 2; ++th) {
            int nt = g * 128 + (Hw >> 4) + th;
            v16bf bm = *(const v16bf*)(wr_t + ((size_t)(nt * 8 + kb) * 32 + lane) * 16);
            acc[g * 2 + th] = __builtin_amdgcn_wmma_f32_16x16x32_bf16(
                false, a, false, bm, (short)0, acc[g * 2 + th], false, false);
          }
      }

      // ---- gates (bias folded here) + cell update in WMMA C layout ----
#pragma unroll
      for (int th = 0; th < 2; ++th) {
        int col = hw_local + th * 16 + nlo;
#pragma unroll
        for (int v = 0; v < 8; ++v) {
          int M = v + rowsel;
          float ig = 1.f / (1.f + __expf(-(acc[0 * 2 + th][v] + bv[0 * 2 + th])));
          float fg = 1.f / (1.f + __expf(-(acc[1 * 2 + th][v] + bv[1 * 2 + th])));
          float gg = tanhf(acc[2 * 2 + th][v] + bv[2 * 2 + th]);
          float og = 1.f / (1.f + __expf(-(acc[3 * 2 + th][v] + bv[3 * 2 + th])));
          float cold = cst[th][v];
          float cnew = fg * cold + ig * gg;
          cnew = fminf(CELL_CLIP, fmaxf(-CELL_CLIP, cnew));
          float mt = (M < 8) ? m_lds[M] : 1.f;
          cnew = (mt > 0.f) ? cnew : cold;
          cst[th][v] = cnew;
          if (M < 8) ht_f32[M * 256 + col] = og * tanhf(cnew);
        }
      }
      __syncthreads();

      // ---- build bf16 A tiles of ht once per WG ----
      fill_a_from_lds(hta, ht_f32);
      __syncthreads();

      // ---- partial projection (ht chunk)@(Wp K-slice) -> hpart[parity][d][wg] ----
      {
        v8f pacc[2];
#pragma unroll
        for (int th = 0; th < 2; ++th)
#pragma unroll
          for (int v = 0; v < 8; ++v) pacc[th][v] = 0.f;
        for (int kb = 0; kb < 8; ++kb) {
          v16bf a = *(const v16bf*)(hta + (kb * 32 + lane) * 16);
#pragma unroll
          for (int th = 0; th < 2; ++th) {
            int nt  = w * 2 + th;          // 16 n-tiles of P=256 split over 8 waves
            int kbg = wg * 8 + kb;         // this WG's K rows of Wp
            v16bf bm = *(const v16bf*)(wp_t + ((size_t)(nt * 64 + kbg) * 32 + lane) * 16);
            pacc[th] = __builtin_amdgcn_wmma_f32_16x16x32_bf16(
                false, a, false, bm, (short)0, pacc[th], false, false);
          }
        }
        int p = sidx & 1;
        float* hp = hpart + (size_t)((p * 2 + d) * 8 + wg) * 4096;
#pragma unroll
        for (int th = 0; th < 2; ++th) {
          int col = hw_local + th * 16 + nlo;
#pragma unroll
          for (int v = 0; v < 8; ++v) {
            int M = v + rowsel;
            hp[M * 256 + col] = pacc[th][v];
          }
        }
      }

      // ---- one grid sync per step (parity double-buffer makes it sufficient) ----
      ++epoch;
      grid_sync(cnt + d, epoch * NWG);

      // ---- reduce K-partials, clip, mask, commit h, store output ----
      {
        int p = sidx & 1;
        const float* hpb = hpart + (size_t)((p * 2 + d) * 8) * 4096;
        for (int i = tid; i < 2048; i += 256) {
          float s = 0.f;
#pragma unroll
          for (int g2 = 0; g2 < 8; ++g2) s += hpb[(size_t)g2 * 4096 + i];
          s = fminf(PROJ_CLIP, fmaxf(-PROJ_CLIP, s));
          int r = i >> 8, col = i & 255;
          float hn = (m_lds[r] > 0.f) ? s : h_f32[i];
          h_f32[i] = hn;
          if (wg == 0) Oout[((size_t)r * T_ + tt) * P_ + col] = hn;
        }
      }
      __syncthreads();

      // ---- rebuild bf16 A tiles of h for next step ----
      fill_a_from_lds(ha, h_f32);
      __syncthreads();
      ++sidx;
    }
    // layer boundary: make Oout globally visible before it is re-read as input
    ++epoch;
    grid_sync(cnt + d, epoch * NWG);
  }
}

// ---------------- ELMo layer value accessor ----------------
__device__ inline float elmo_val(int ly, int b, int t, int c,
                                 const float* __restrict__ emb,
                                 const float* __restrict__ O) {
  if (ly == 0) return emb[((size_t)b * T_ + t) * E_ + (c & 255)];
  int dir = (c < 256) ? 0 : 1;
  int cc = c & 255;
  float v0 = O[(size_t)(dir * 2 + 0) * (B_ * T_ * P_) + ((size_t)b * T_ + t) * P_ + cc];
  if (ly == 1) return v0;
  float v1 = O[(size_t)(dir * 2 + 1) * (B_ * T_ * P_) + ((size_t)b * T_ + t) * P_ + cc];
  return v0 + v1;
}

// ---------------- masked mean/var per (batch, layer) ----------------
__global__ void stats_kernel(const float* __restrict__ mask,
                             const float* __restrict__ emb,
                             const float* __restrict__ O,
                             float* __restrict__ stats) {  // [B][3][2]
  const int b = blockIdx.x, ly = blockIdx.y, tid = threadIdx.x;
  __shared__ float red[256];
  float num = 0.f;
  for (int t = 0; t < T_; ++t) num += mask[b * T_ + t];
  float s = 0.f;
  for (int i = tid; i < T_ * 512; i += 256) {
    int t = i >> 9, c = i & 511;
    s += elmo_val(ly, b, t, c, emb, O) * mask[b * T_ + t];
  }
  red[tid] = s; __syncthreads();
  for (int off = 128; off > 0; off >>= 1) { if (tid < off) red[tid] += red[tid + off]; __syncthreads(); }
  float mean = red[0] / num;
  __syncthreads();
  float s2 = 0.f;
  for (int i = tid; i < T_ * 512; i += 256) {
    int t = i >> 9, c = i & 511;
    float m = mask[b * T_ + t];
    float dd = (elmo_val(ly, b, t, c, emb, O) * m - mean) * m;
    s2 += dd * dd;
  }
  red[tid] = s2; __syncthreads();
  for (int off = 128; off > 0; off >>= 1) { if (tid < off) red[tid] += red[tid + off]; __syncthreads(); }
  if (tid == 0) { stats[(b * 3 + ly) * 2 + 0] = mean; stats[(b * 3 + ly) * 2 + 1] = red[0] / num; }
}

// ---------------- softmax-weighted combine ----------------
__global__ void combine_kernel(const float* __restrict__ emb, const float* __restrict__ O,
                               const float* __restrict__ stats,
                               const float* __restrict__ elmo_w, const float* __restrict__ gamma,
                               float* __restrict__ out) {
  long idx = (long)blockIdx.x * 256 + threadIdx.x;
  if (idx >= (long)B_ * T_ * 512) return;
  int c = (int)(idx & 511);
  int t = (int)((idx >> 9) % T_);
  int b = (int)(idx / ((long)512 * T_));
  float w0 = elmo_w[0], w1 = elmo_w[1], w2 = elmo_w[2];
  float mx = fmaxf(w0, fmaxf(w1, w2));
  float e0 = __expf(w0 - mx), e1 = __expf(w1 - mx), e2 = __expf(w2 - mx);
  float inv = 1.f / (e0 + e1 + e2);
  float wsm[3] = {e0 * inv, e1 * inv, e2 * inv};
  float acc = 0.f;
#pragma unroll
  for (int ly = 0; ly < 3; ++ly) {
    float mean = stats[(b * 3 + ly) * 2 + 0];
    float var  = stats[(b * 3 + ly) * 2 + 1];
    acc += wsm[ly] * (elmo_val(ly, b, t, c, emb, O) - mean) * rsqrtf(var + 1e-12f);
  }
  out[idx] = acc * gamma[0];
}

// ---------------- host launcher ----------------
extern "C" void kernel_launch(void* const* d_in, const int* in_sizes, int n_in,
                              void* d_out, int out_size, void* d_ws, size_t ws_size,
                              hipStream_t stream) {
  (void)in_sizes; (void)n_in; (void)out_size; (void)ws_size;
  const int*   tokens = (const int*)d_in[0];
  const float* mask   = (const float*)d_in[1];
  const float* table  = (const float*)d_in[2];
  const float* Wk     = (const float*)d_in[3];
  const float* Wr     = (const float*)d_in[4];
  const float* bias   = (const float*)d_in[5];
  const float* Wp     = (const float*)d_in[6];
  const float* elmo_w = (const float*)d_in[7];
  const float* gamma  = (const float*)d_in[8];
  float* out = (float*)d_out;
  char*  ws  = (char*)d_ws;

  const size_t o_cnt = 0;
  const size_t o_emb = 256;
  const size_t o_wkt = o_emb + (size_t)B_ * T_ * E_ * 4;          // 786 KB
  const size_t o_wrt = o_wkt + (size_t)4 * 256 * 8192 * 2;        // +16 MB
  const size_t o_wpt = o_wrt + (size_t)4 * 256 * 8192 * 2;        // +16 MB
  const size_t o_O   = o_wpt + (size_t)4 * 2048 * 256 * 2;        // +4 MB
  const size_t o_hp  = o_O   + (size_t)4 * B_ * T_ * P_ * 4;      // +3 MB
  const size_t o_st  = o_hp  + (size_t)32 * 16 * 256 * 4;         // +512 KB

  unsigned* cnt   = (unsigned*)(ws + o_cnt);
  float*    emb   = (float*)(ws + o_emb);
  __bf16*   wkt   = (__bf16*)(ws + o_wkt);
  __bf16*   wrt   = (__bf16*)(ws + o_wrt);
  __bf16*   wpt   = (__bf16*)(ws + o_wpt);
  float*    Obuf  = (float*)(ws + o_O);
  float*    hpart = (float*)(ws + o_hp);
  float*    stats = (float*)(ws + o_st);

  hipMemsetAsync(cnt, 0, 2 * sizeof(unsigned), stream);

  gather_kernel<<<(B_ * T_ * E_) / 256, 256, 0, stream>>>(tokens, table, emb);

  const long nWk = (long)4 * 256 * 8192;
  pack_w_kernel<<<(int)((nWk + 255) / 256), 256, 0, stream>>>(Wk, wkt, 256, 8192, nWk);
  pack_w_kernel<<<(int)((nWk + 255) / 256), 256, 0, stream>>>(Wr, wrt, 256, 8192, nWk);
  const long nWp = (long)4 * 2048 * 256;
  pack_w_kernel<<<(int)((nWp + 255) / 256), 256, 0, stream>>>(Wp, wpt, 2048, 256, nWp);

  lstm_kernel<<<dim3(NWG, 2), 256, 0, stream>>>(emb, wkt, wrt, wpt, bias, mask,
                                                Obuf, hpart, cnt);

  stats_kernel<<<dim3(B_, 3), 256, 0, stream>>>(mask, emb, Obuf, stats);

  const long nOut = (long)B_ * T_ * 512;
  combine_kernel<<<(int)((nOut + 255) / 256), 256, 0, stream>>>(emb, Obuf, stats,
                                                                elmo_w, gamma, out);
}
